// ProxyMemory_24283745091969
// MI455X (gfx1250) — compile-verified, matrix operand order checked by reading, not run on hardware
//
#include <hip/hip_runtime.h>
#include <hip/hip_bf16.h>

// ---------------- problem constants (match reference) ----------------
#define BATCH   256
#define DIM     256
#define NPROXY  100000
#define NEG_K   50
#define INV_TEMP 20.0f           // 1 / 0.05

#define COLS_PER_BLOCK 80        // proxy columns per block (5 N-tiles); 100000/80 = 1250 exact
#define NTILES (COLS_PER_BLOCK / 16)
#define LDS_ROW_H (DIM + 8)      // padded row stride in halves (264) -> bank-friendly

// workspace layout (bytes)
#define SCORES_BYTES ((size_t)BATCH * (size_t)NPROXY * 4)   // 102,400,000
#define FEATH_BYTES  ((size_t)BATCH * (size_t)DIM * 2)      // 131,072
#define FEATH_OFF    SCORES_BYTES
#define BLBL_OFF     (SCORES_BYTES + FEATH_BYTES)

typedef __attribute__((ext_vector_type(16))) _Float16 v16h;
typedef __attribute__((ext_vector_type(8)))  _Float16 v8h;
typedef __attribute__((ext_vector_type(8)))  float    v8f;
typedef __attribute__((ext_vector_type(4)))  float    v4f;

union HFrag { v16h v; v8h h[2]; };

// ------------------------------------------------------------------
// Kernel 1: per-sample pseudo-label gather + features f32->f16 + zero out
// ------------------------------------------------------------------
__global__ void prep_kernel(const float* __restrict__ features,
                            const int*   __restrict__ index_labels,
                            const int*   __restrict__ img_proxy_index,
                            const int*   __restrict__ all_proxy_label,
                            _Float16*    __restrict__ featH,
                            int*         __restrict__ batch_lbl,
                            float*       __restrict__ out) {
  int b = blockIdx.x;
  int t = threadIdx.x;
  featH[b * DIM + t] = (_Float16)features[b * DIM + t];
  if (b == 0) {
    int pidx = img_proxy_index[index_labels[t]];
    batch_lbl[t] = all_proxy_label[pidx];
    if (t == 0) out[0] = 0.0f;   // d_out is poisoned; we accumulate into it
  }
}

// ------------------------------------------------------------------
// Kernel 2: scores[256][100000] = (featH @ proxy^T) * INV_TEMP  via WMMA.
// grid.x = 1250 blocks of 80 proxy columns.  block = 256 threads = 8 waves.
//
// Phase 1: 256 threads stage 80 proxy rows (f32 -> f16) into padded LDS.
// Phase 2: wave w owns M-tiles 2w, 2w+1 and loops the block's 5 N-tiles.
//   Per K-step: load A (4 global b128) and ALL 5 B fragments (10 ds b128,
//   distinct registers) up front, then fire 10 WMMAs — lets the compiler
//   clause the ds loads and drain with descending s_wait_dscnt instead of
//   a serialized load->wait(0)->wmma chain per N-tile.
//
// Fragment lane layout (16-bit A 16x32, ISA 7.12.2), mirrored for B:
//   lane l: row/col = l%16 ; halves[0..7]  = K + koff + 0..7,
//                            halves[8..15] = K + 16 + koff + 0..7,
//   koff = 0 for lanes 0..15, 8 for lanes 16..31.
// ------------------------------------------------------------------
__global__ void gemm_kernel(const _Float16* __restrict__ featH,
                            const float*    __restrict__ proxy,
                            float*          __restrict__ scores) {
  __shared__ __align__(16) _Float16 bsh[COLS_PER_BLOCK * LDS_ROW_H]; // 42,240 B

  // ---- phase 1: stage 80 proxy rows -> LDS as f16 ----
  {
    const int t = threadIdx.x;
    const size_t pbase = (size_t)blockIdx.x * COLS_PER_BLOCK;
#pragma unroll
    for (int j = 0; j < 5; ++j) {
      const int c   = t + j * 256;        // chunk id: 0..1279
      const int r   = c >> 4;             // proxy row within block: 0..79
      const int col = (c & 15) * 16;      // K offset: 0..240
      const float* __restrict__ src = proxy + (pbase + r) * DIM + col;
      if (j == 0) __builtin_prefetch(src, 0, 0);   // global_prefetch_b8
      _Float16* dst = bsh + r * LDS_ROW_H + col;
      v4f f0 = *(const v4f*)(src);
      v4f f1 = *(const v4f*)(src + 4);
      v4f f2 = *(const v4f*)(src + 8);
      v4f f3 = *(const v4f*)(src + 12);
      v8h h0, h1;
#pragma unroll
      for (int k = 0; k < 4; ++k) {
        h0[k] = (_Float16)f0[k]; h0[k + 4] = (_Float16)f1[k];
        h1[k] = (_Float16)f2[k]; h1[k + 4] = (_Float16)f3[k];
      }
      *(v8h*)(dst)     = h0;
      *(v8h*)(dst + 8) = h1;
    }
  }
  __syncthreads();

  // ---- phase 2: WMMA ----
  const int lane = threadIdx.x & 31;
  const int wave = threadIdx.x >> 5;
  const int lm   = lane & 15;
  const int koff = (lane < 16) ? 0 : 8;

  const int m0 = wave * 32 + lm;      // M-tile 0 row for this lane
  const int m1 = m0 + 16;             // M-tile 1 row

  const _Float16* __restrict__ bbase = bsh + lm * LDS_ROW_H;

  v8f acc0[NTILES];   // M-tile 0 accumulators
  v8f acc1[NTILES];   // M-tile 1 accumulators
#pragma unroll
  for (int nt = 0; nt < NTILES; ++nt) { acc0[nt] = (v8f){}; acc1[nt] = (v8f){}; }

#pragma unroll
  for (int kt = 0; kt < DIM / 32; ++kt) {
    const int kb = kt * 32 + koff;

    HFrag a0, a1;
    a0.h[0] = *(const v8h*)(featH + m0 * DIM + kb);
    a0.h[1] = *(const v8h*)(featH + m0 * DIM + kb + 16);
    a1.h[0] = *(const v8h*)(featH + m1 * DIM + kb);
    a1.h[1] = *(const v8h*)(featH + m1 * DIM + kb + 16);

    // load ALL B fragments of this K-step into distinct registers first
    HFrag bf[NTILES];
#pragma unroll
    for (int nt = 0; nt < NTILES; ++nt) {
      bf[nt].h[0] = *(const v8h*)(bbase + nt * 16 * LDS_ROW_H + kb);
      bf[nt].h[1] = *(const v8h*)(bbase + nt * 16 * LDS_ROW_H + kb + 16);
    }

#pragma unroll
    for (int nt = 0; nt < NTILES; ++nt) {
      acc0[nt] = __builtin_amdgcn_wmma_f32_16x16x32_f16(false, a0.v, false, bf[nt].v,
                                                        (short)0, acc0[nt], false, false);
      acc1[nt] = __builtin_amdgcn_wmma_f32_16x16x32_f16(false, a1.v, false, bf[nt].v,
                                                        (short)0, acc1[nt], false, false);
    }
  }

  // C/D layout: VGPR r -> M = r (lanes 0..15) or 8+r (lanes 16..31), N = lane&15.
  const int rhi = (lane < 16) ? 0 : 8;
  const size_t pbase = (size_t)blockIdx.x * COLS_PER_BLOCK + lm;
#pragma unroll
  for (int nt = 0; nt < NTILES; ++nt) {
    const size_t p = pbase + nt * 16;
#pragma unroll
    for (int r = 0; r < 8; ++r) {
      const size_t row0 = (size_t)(wave * 32 + rhi + r) * NPROXY;
      const size_t row1 = (size_t)(wave * 32 + 16 + rhi + r) * NPROXY;
      scores[row0 + p] = acc0[nt][r] * INV_TEMP;
      scores[row1 + p] = acc1[nt][r] * INV_TEMP;
    }
  }
}

// ------------------------------------------------------------------
// Kernel 3: per-row selection + loss.
// grid.x = 256 rows, block = 64 threads.
// loss_i = (min(npos,50)/npos) * LSE(sel) - sum(pos scores)/npos
// sel = positives U top-(50 - npos) negatives (softmax order-invariant).
// ------------------------------------------------------------------
__global__ void topk_loss_kernel(const float* __restrict__ scores,
                                 const int*   __restrict__ all_proxy_label,
                                 const int*   __restrict__ batch_lbl,
                                 float*       __restrict__ out) {
  const int row = blockIdx.x;
  const int tid = threadIdx.x;
  const int lbl = batch_lbl[row];
  const float* __restrict__ srow = scores + (size_t)row * NPROXY;

  float nt[NEG_K];
#pragma unroll
  for (int i = 0; i < NEG_K; ++i) nt[i] = -1e30f;

  int   cnt = 0;
  float sum = 0.0f;
  float M = -1e30f, S = 0.0f;        // streaming log-sum-exp of positives

  for (int pidx = tid; pidx < NPROXY; pidx += 64) {
    float v = srow[pidx];
    if (all_proxy_label[pidx] == lbl) {
      ++cnt;
      sum += v;
      if (v > M) { S = S * __expf(M - v) + 1.0f; M = v; }
      else       { S += __expf(v - M); }
    } else if (v > nt[NEG_K - 1]) {
      int j = NEG_K - 1;
      while (j > 0 && nt[j - 1] < v) { nt[j] = nt[j - 1]; --j; }
      nt[j] = v;
    }
  }

  __shared__ float lnt[64][NEG_K];
  __shared__ float lm[64], ls[64], lsum[64];
  __shared__ int   lcnt[64];
#pragma unroll
  for (int i = 0; i < NEG_K; ++i) lnt[tid][i] = nt[i];
  lm[tid] = M; ls[tid] = S; lsum[tid] = sum; lcnt[tid] = cnt;
  __syncthreads();

  if (tid == 0) {
    int   tcnt = 0;
    float tsum = 0.0f;
    float TM = -1e30f, TS = 0.0f;
    for (int t = 0; t < 64; ++t) {
      tcnt += lcnt[t];
      tsum += lsum[t];
      float m2 = lm[t], s2 = ls[t];
      if (m2 > TM) { TS = TS * __expf(TM - m2) + s2; TM = m2; }
      else         { TS += s2 * __expf(m2 - TM); }
    }

    int head[64];
    for (int t = 0; t < 64; ++t) head[t] = 0;
    int sel_pos = (tcnt < NEG_K) ? tcnt : NEG_K;
    int need = NEG_K - sel_pos;
    for (int k = 0; k < need; ++k) {
      float best = -1e30f; int bt = 0;
      for (int t = 0; t < 64; ++t) {
        if (head[t] < NEG_K) {
          float v = lnt[t][head[t]];
          if (v > best) { best = v; bt = t; }
        }
      }
      head[bt]++;
      if (best > TM) { TS = TS * __expf(TM - best) + 1.0f; TM = best; }
      else           { TS += __expf(best - TM); }
    }

    float npos   = fmaxf((float)tcnt, 1.0f);
    float lse    = TM + __logf(TS);
    float mass   = (float)sel_pos / npos;
    float loss_i = mass * lse - tsum / npos;
    atomicAdd(out, loss_i * (1.0f / (float)BATCH));
  }
}

// ------------------------------------------------------------------
extern "C" void kernel_launch(void* const* d_in, const int* in_sizes, int n_in,
                              void* d_out, int out_size, void* d_ws, size_t ws_size,
                              hipStream_t stream) {
  const float* features        = (const float*)d_in[0];
  const int*   index_labels    = (const int*)  d_in[1];
  const float* proxy_memory    = (const float*)d_in[2];
  const int*   img_proxy_index = (const int*)  d_in[3];
  const int*   all_proxy_label = (const int*)  d_in[4];
  float*       out             = (float*)d_out;

  char*     ws        = (char*)d_ws;
  float*    scores    = (float*)ws;
  _Float16* featH     = (_Float16*)(ws + FEATH_OFF);
  int*      batch_lbl = (int*)(ws + BLBL_OFF);

  prep_kernel<<<BATCH, DIM, 0, stream>>>(features, index_labels, img_proxy_index,
                                         all_proxy_label, featH, batch_lbl, out);
  gemm_kernel<<<NPROXY / COLS_PER_BLOCK, 256, 0, stream>>>(featH, proxy_memory, scores);
  topk_loss_kernel<<<BATCH, 64, 0, stream>>>(scores, all_proxy_label, batch_lbl, out);
}
